// CausalSelfAttention_26474178413207
// MI455X (gfx1250) — matrix operand order repaired and checked by reading, not compiled, not run on hardware
//
#include <hip/hip_runtime.h>

// ---------------- problem constants ----------------
#define DD   1024      // model dim
#define TT   2048      // seq len
#define BB   2         // batch
#define NTOK (BB*TT)   // 4096 tokens
#define NH   16        // heads
#define HDM  64        // head dim

typedef __attribute__((ext_vector_type(16))) __bf16 v16bf;
typedef __attribute__((ext_vector_type(8)))  float  v8f;
typedef __attribute__((ext_vector_type(4)))  unsigned int v4u;

union FragBF { v16bf v; v4u u[2]; };

__device__ inline unsigned short f2bf(float f) {
    union { float f; unsigned int u; } x; x.f = f;
    unsigned int r = x.u + 0x7FFFu + ((x.u >> 16) & 1u);   // RNE
    return (unsigned short)(r >> 16);
}

__device__ inline v8f zero_v8f() {
    v8f z;
#pragma unroll
    for (int e = 0; e < 8; ++e) z[e] = 0.0f;
    return z;
}

__device__ inline v8f wmma_bf16(v16bf a, v16bf b, v8f c) {
    return __builtin_amdgcn_wmma_f32_16x16x32_bf16(
        /*neg_a=*/false, a, /*neg_b=*/false, b,
        /*c_mod=*/(short)0, c, /*reuse_a=*/false, /*reuse_b=*/false);
}

// A fragment (16x32 bf16, M rows x K): lane = h*16+m holds row m,
// elems 0..7 -> K = h*8+j , elems 8..15 -> K = 16+h*8+j  (ISA 7.12.2)
__device__ inline v16bf load_a_frag(const unsigned short* base, int ld, int lane) {
    int h = lane >> 4, m = lane & 15;
    const unsigned short* p = base + m * ld;
    FragBF f;
    f.u[0] = *(const v4u*)(p + h * 8);
    f.u[1] = *(const v4u*)(p + 16 + h * 8);
    return f.v;
}

// B fragment (K x 16 bf16) where column n of B is a contiguous row of the
// row-major source S:  B[k][n] = S[n][k].  Lane = h*16+n holds column n,
// elems 0..7 -> K = h*8+j , elems 8..15 -> K = 16+h*8+j  (same as A).
__device__ inline v16bf load_b_frag(const unsigned short* base, int ld, int lane) {
    int h = lane >> 4, n = lane & 15;
    const unsigned short* p = base + n * ld;
    FragBF f;
    f.u[0] = *(const v4u*)(p + h * 8);
    f.u[1] = *(const v4u*)(p + 16 + h * 8);
    return f.v;
}

__device__ inline void wave_lds_fence() {
    __builtin_amdgcn_wave_barrier();
    asm volatile("s_wait_dscnt 0x0" ::: "memory");
    __builtin_amdgcn_wave_barrier();
}

// Async DMA of 16 bytes from global to LDS (ASYNCcnt path, gfx1250).
__device__ inline void async_b128(unsigned lds_byte_addr, const void* gptr) {
    asm volatile("global_load_async_to_lds_b128 %0, %1, off"
                 :: "v"(lds_byte_addr), "v"((unsigned long long)gptr)
                 : "memory");
}

// ---------------- stage 0: fp32 -> bf16 ----------------
__global__ void cvt_f32_bf16(const float* __restrict__ src,
                             unsigned short* __restrict__ dst, int n) {
    int i = blockIdx.x * blockDim.x + threadIdx.x;
    if (i < n) dst[i] = f2bf(src[i]);
}

// ---------------- shared 128x128 GEMM body ----------------
// Block = 256 threads (8 waves).  C(128x128) = A(128xK) * B(128xK)^T with
// both operands staged in LDS through async global->LDS DMA, double buffered.
// Wave wid: mslot = (wid>>1)*32 rows, nslot = (wid&1)*64 cols; 2x4 WMMA tiles.
#define KSTEP 32
#define LDW   (KSTEP)          // LDS tile leading dim (elements)

__device__ inline void issue_tile(unsigned short* lbuf,
                                  const unsigned short* g, int ld,
                                  int row, int half, int k0) {
    const unsigned short* gp = g + row * ld + k0 + half * 16;
    unsigned loff = (unsigned)(unsigned long long)(lbuf + row * LDW + half * 16);
    async_b128(loff,      gp);
    async_b128(loff + 16, gp + 8);
}

__device__ inline void gemm_128x128(const unsigned short* Ag,   // 128 rows, ld=DD
                                    const unsigned short* Bg,   // 128 rows, ld=DD
                                    unsigned short (*Abuf)[128 * LDW],
                                    unsigned short (*Bbuf)[128 * LDW],
                                    v8f acc[2][4]) {
    int tid  = threadIdx.x;
    int lane = tid & 31;
    int wid  = tid >> 5;
    int mslot = (wid >> 1) * 32;
    int nslot = (wid & 1) * 64;
    int row  = tid >> 1;       // 0..127
    int half = tid & 1;        // which 32B chunk of the 64B row

#pragma unroll
    for (int i = 0; i < 2; ++i)
#pragma unroll
        for (int j = 0; j < 4; ++j) acc[i][j] = zero_v8f();

    // prologue: fill buffer 0
    issue_tile(Abuf[0], Ag, DD, row, half, 0);
    issue_tile(Bbuf[0], Bg, DD, row, half, 0);

    for (int k0 = 0; k0 < DD; k0 += KSTEP) {
        int cur = (k0 >> 5) & 1;
        if (k0 + KSTEP < DD) {
            issue_tile(Abuf[cur ^ 1], Ag, DD, row, half, k0 + KSTEP);
            issue_tile(Bbuf[cur ^ 1], Bg, DD, row, half, k0 + KSTEP);
            asm volatile("s_wait_asynccnt 0x4" ::: "memory");  // prev 4 done
        } else {
            asm volatile("s_wait_asynccnt 0x0" ::: "memory");
        }
        __syncthreads();                       // tile visible to all waves

        v16bf a[2], b[4];
#pragma unroll
        for (int i = 0; i < 2; ++i)
            a[i] = load_a_frag(Abuf[cur] + (mslot + i * 16) * LDW, LDW, lane);
#pragma unroll
        for (int j = 0; j < 4; ++j)
            b[j] = load_b_frag(Bbuf[cur] + (nslot + j * 16) * LDW, LDW, lane);
#pragma unroll
        for (int i = 0; i < 2; ++i)
#pragma unroll
            for (int j = 0; j < 4; ++j)
                acc[i][j] = wmma_bf16(a[i], b[j], acc[i][j]);

        __syncthreads();                       // safe to overwrite buffer
    }
}

// ---------------- stage 1: QKV projection ----------------
// which = 0:Q (scaled 1/8, [B,H,T,HD])  1:K ([B,H,T,HD])  2:V (transposed [B,H,HD,T])
__global__ __launch_bounds__(256) void proj_qkv(
    const unsigned short* __restrict__ xb,
    const unsigned short* __restrict__ wqb,
    const unsigned short* __restrict__ wkb,
    const unsigned short* __restrict__ wvb,
    const float* __restrict__ bq, const float* __restrict__ bk,
    const float* __restrict__ bv,
    unsigned short* __restrict__ Qb, unsigned short* __restrict__ Kb,
    unsigned short* __restrict__ Vt)
{
    __shared__ __align__(16) unsigned short Abuf[2][128 * LDW];
    __shared__ __align__(16) unsigned short Bbuf[2][128 * LDW];

    int bid   = blockIdx.x;
    int which = bid >> 8;                 // 256 tiles per projection
    int rem   = bid & 255;
    int m0 = (rem >> 3) * 128;            // token tile (32 m-tiles)
    int n0 = (rem & 7) * 128;             // out-dim tile (8 n-tiles)

    const unsigned short* W = which == 0 ? wqb : (which == 1 ? wkb : wvb);
    const float* bias       = which == 0 ? bq  : (which == 1 ? bk  : bv);

    v8f acc[2][4];
    gemm_128x128(xb + (size_t)m0 * DD, W + (size_t)n0 * DD, Abuf, Bbuf, acc);

    int lane = threadIdx.x & 31, wid = threadIdx.x >> 5;
    int mslot = (wid >> 1) * 32, nslot = (wid & 1) * 64;
    int h = lane >> 4, c = lane & 15;
#pragma unroll
    for (int i = 0; i < 2; ++i)
#pragma unroll
        for (int j = 0; j < 4; ++j)
#pragma unroll
            for (int e = 0; e < 8; ++e) {
                int tok = m0 + mslot + i * 16 + h * 8 + e;
                int col = n0 + nslot + j * 16 + c;
                float v = acc[i][j][e] + bias[col];
                int bb = tok >> 11, tl = tok & (TT - 1);
                int head = col >> 6, hd = col & (HDM - 1);
                if (which == 0) {
                    v *= 0.125f;  // 1/sqrt(64) folded into Q
                    Qb[(((bb * NH + head) * TT + tl) << 6) + hd] = f2bf(v);
                } else if (which == 1) {
                    Kb[(((bb * NH + head) * TT + tl) << 6) + hd] = f2bf(v);
                } else {
                    Vt[((bb * NH + head) * HDM + hd) * TT + tl] = f2bf(v);
                }
            }
}

// ---------------- stage 2: causal flash attention ----------------
// One wave owns a 16-row q tile of one (b,head); streams 32-wide k chunks.
__global__ __launch_bounds__(128) void attn(
    const unsigned short* __restrict__ Qb,
    const unsigned short* __restrict__ Kb,
    const unsigned short* __restrict__ Vt,
    unsigned short* __restrict__ Yb)
{
    __shared__ __align__(16) unsigned short pbuf[4][16 * 32];

    int wid  = threadIdx.x >> 5;
    int lane = threadIdx.x & 31;
    int gw   = blockIdx.x * 4 + wid;
    int b    = gw >> 11;                 // 2048 waves per batch (16 heads * 128 qtiles)
    int rem  = gw & 2047;
    int head = rem >> 7;
    int qt0  = (rem & 127) * 16;

    const unsigned short* Qh = Qb + (b * NH + head) * TT * HDM;
    const unsigned short* Kh = Kb + (b * NH + head) * TT * HDM;
    const unsigned short* Vh = Vt + (b * NH + head) * HDM * TT;

    int h = lane >> 4, c = lane & 15;

    v16bf aq0 = load_a_frag(Qh + qt0 * HDM + 0,  HDM, lane);
    v16bf aq1 = load_a_frag(Qh + qt0 * HDM + 32, HDM, lane);

    v8f o[4];
#pragma unroll
    for (int j = 0; j < 4; ++j) o[j] = zero_v8f();
    float mrow[8], lrow[8];
#pragma unroll
    for (int e = 0; e < 8; ++e) { mrow[e] = -INFINITY; lrow[e] = 0.0f; }

    const float LOG2E = 1.44269504f;
    int kend = qt0 + 16;                         // last valid k is qt0+15
    for (int kc0 = 0; kc0 < kend; kc0 += 32) {
        // S = Q K^T : two 16-col tiles, K-dim 64 = 2 bf16 WMMAs each
        v8f s0 = zero_v8f(), s1 = zero_v8f();
        s0 = wmma_bf16(aq0, load_b_frag(Kh + (kc0 +  0) * HDM +  0, HDM, lane), s0);
        s0 = wmma_bf16(aq1, load_b_frag(Kh + (kc0 +  0) * HDM + 32, HDM, lane), s0);
        s1 = wmma_bf16(aq0, load_b_frag(Kh + (kc0 + 16) * HDM +  0, HDM, lane), s1);
        s1 = wmma_bf16(aq1, load_b_frag(Kh + (kc0 + 16) * HDM + 32, HDM, lane), s1);

        float p0[8], p1[8], alpha[8];
#pragma unroll
        for (int e = 0; e < 8; ++e) {
            int q = qt0 + h * 8 + e;
            float v0 = (kc0 + c)      <= q ? s0[e] : -INFINITY;   // causal mask
            float v1 = (kc0 + 16 + c) <= q ? s1[e] : -INFINITY;
            float rmax = fmaxf(v0, v1);
            rmax = fmaxf(rmax, __shfl_xor(rmax, 1));
            rmax = fmaxf(rmax, __shfl_xor(rmax, 2));
            rmax = fmaxf(rmax, __shfl_xor(rmax, 4));
            rmax = fmaxf(rmax, __shfl_xor(rmax, 8));
            float mnew = fmaxf(mrow[e], rmax);
            alpha[e] = exp2f((mrow[e] - mnew) * LOG2E);
            p0[e]    = exp2f((v0 - mnew) * LOG2E);
            p1[e]    = exp2f((v1 - mnew) * LOG2E);
            float rs = p0[e] + p1[e];
            rs += __shfl_xor(rs, 1);
            rs += __shfl_xor(rs, 2);
            rs += __shfl_xor(rs, 4);
            rs += __shfl_xor(rs, 8);
            lrow[e] = lrow[e] * alpha[e] + rs;
            mrow[e] = mnew;
        }
#pragma unroll
        for (int j = 0; j < 4; ++j)
#pragma unroll
            for (int e = 0; e < 8; ++e) o[j][e] *= alpha[e];

        // P (C-layout, f32) -> bf16 A-fragment via LDS transpose
        unsigned short* pb = pbuf[wid];
#pragma unroll
        for (int e = 0; e < 8; ++e) {
            int r = h * 8 + e;
            pb[r * 32 + c]      = f2bf(p0[e]);
            pb[r * 32 + 16 + c] = f2bf(p1[e]);
        }
        wave_lds_fence();
        FragBF pf;
        {
            const unsigned short* p = pb + (lane & 15) * 32;
            pf.u[0] = *(const v4u*)(p + h * 8);
            pf.u[1] = *(const v4u*)(p + 16 + h * 8);
        }
        // O += P V  (V stored transposed: columns contiguous)
#pragma unroll
        for (int j = 0; j < 4; ++j) {
            v16bf bvf = load_b_frag(Vh + (j * 16) * TT + kc0, TT, lane);
            o[j] = wmma_bf16(pf.v, bvf, o[j]);
        }
    }

#pragma unroll
    for (int e = 0; e < 8; ++e) lrow[e] = 1.0f / lrow[e];
#pragma unroll
    for (int j = 0; j < 4; ++j)
#pragma unroll
        for (int e = 0; e < 8; ++e) {
            int r   = h * 8 + e;
            int tok = b * TT + qt0 + r;
            int col = head * HDM + j * 16 + c;
            Yb[tok * DD + col] = f2bf(o[j][e] * lrow[e]);
        }
}

// ---------------- stage 3: output projection (fp32 out) ----------------
__global__ __launch_bounds__(256) void proj_out(
    const unsigned short* __restrict__ yb,
    const unsigned short* __restrict__ wob,
    const float* __restrict__ bo, float* __restrict__ out)
{
    __shared__ __align__(16) unsigned short Abuf[2][128 * LDW];
    __shared__ __align__(16) unsigned short Bbuf[2][128 * LDW];

    int rem = blockIdx.x;                 // 256 tiles
    int m0 = (rem >> 3) * 128;
    int n0 = (rem & 7) * 128;

    v8f acc[2][4];
    gemm_128x128(yb + (size_t)m0 * DD, wob + (size_t)n0 * DD, Abuf, Bbuf, acc);

    int lane = threadIdx.x & 31, wid = threadIdx.x >> 5;
    int mslot = (wid >> 1) * 32, nslot = (wid & 1) * 64;
    int h = lane >> 4, c = lane & 15;
#pragma unroll
    for (int i = 0; i < 2; ++i)
#pragma unroll
        for (int j = 0; j < 4; ++j)
#pragma unroll
            for (int e = 0; e < 8; ++e) {
                int tok = m0 + mslot + i * 16 + h * 8 + e;
                int col = n0 + nslot + j * 16 + c;
                out[tok * DD + col] = acc[i][j][e] + bo[col];
            }
}

// ---------------- launch ----------------
extern "C" void kernel_launch(void* const* d_in, const int* in_sizes, int n_in,
                              void* d_out, int out_size, void* d_ws, size_t ws_size,
                              hipStream_t stream) {
    const float* x  = (const float*)d_in[0];
    const float* Wq = (const float*)d_in[1];
    const float* bq = (const float*)d_in[2];
    const float* Wk = (const float*)d_in[3];
    const float* bk = (const float*)d_in[4];
    const float* Wv = (const float*)d_in[5];
    const float* bv = (const float*)d_in[6];
    const float* Wo = (const float*)d_in[7];
    const float* bo = (const float*)d_in[8];
    float* out = (float*)d_out;

    char* ws = (char*)d_ws;
    size_t off = 0;
    unsigned short* xb  = (unsigned short*)(ws + off); off += (size_t)NTOK * DD * 2;   // 8 MB
    unsigned short* wqb = (unsigned short*)(ws + off); off += (size_t)DD * DD * 2;     // 2 MB
    unsigned short* wkb = (unsigned short*)(ws + off); off += (size_t)DD * DD * 2;
    unsigned short* wvb = (unsigned short*)(ws + off); off += (size_t)DD * DD * 2;
    unsigned short* wob = (unsigned short*)(ws + off); off += (size_t)DD * DD * 2;
    unsigned short* Qb  = (unsigned short*)(ws + off); off += (size_t)NTOK * DD * 2;   // 8 MB
    unsigned short* Kb  = (unsigned short*)(ws + off); off += (size_t)NTOK * DD * 2;
    unsigned short* Vt  = (unsigned short*)(ws + off); off += (size_t)NTOK * DD * 2;
    unsigned short* Yb  = (unsigned short*)(ws + off); off += (size_t)NTOK * DD * 2;

    int nx = NTOK * DD;   // 4,194,304
    int nw = DD * DD;     // 1,048,576
    cvt_f32_bf16<<<nx / 256, 256, 0, stream>>>(x,  xb,  nx);
    cvt_f32_bf16<<<nw / 256, 256, 0, stream>>>(Wq, wqb, nw);
    cvt_f32_bf16<<<nw / 256, 256, 0, stream>>>(Wk, wkb, nw);
    cvt_f32_bf16<<<nw / 256, 256, 0, stream>>>(Wv, wvb, nw);
    cvt_f32_bf16<<<nw / 256, 256, 0, stream>>>(Wo, wob, nw);

    // 3 projections * (32 x 8) 128x128 tiles = 768 blocks
    proj_qkv<<<768, 256, 0, stream>>>(xb, wqb, wkb, wvb, bq, bk, bv, Qb, Kb, Vt);

    // B*H*(T/16) = 4096 q-tiles, 4 waves/block
    attn<<<1024, 128, 0, stream>>>(Qb, Kb, Vt, Yb);

    // (32 x 8) 128x128 tiles = 256 blocks
    proj_out<<<256, 256, 0, stream>>>(Yb, wob, bo, out);

    (void)in_sizes; (void)n_in; (void)out_size; (void)ws_size;
}
